// L2TransformerConv_84859963834422
// MI455X (gfx1250) — matrix operand order repaired and checked by reading, not compiled, not run on hardware
//
#include <hip/hip_runtime.h>
#include <hip/hip_bf16.h>
#include <stdint.h>

// ---------------- problem constants ----------------
#define N_NODES 50000
#define N_EDGES 800000
#define IN_CH   256
#define HID     200
#define OUT_CH  2

#define MT  3125   // M tiles = 50000/16 exactly
#define NT1 13     // N tiles = ceil(200/16)
#define KS1 8      // K steps = 256/32

typedef __attribute__((ext_vector_type(16))) __bf16 v16bf;
typedef __attribute__((ext_vector_type(8)))  float  v8f;

// ---------------- workspace layout (bytes) ----------------
constexpr size_t A256(size_t x) { return (x + 255) & ~size_t(255); }
constexpr size_t SZ_NH    = A256((size_t)N_NODES * HID * 4);
constexpr size_t SZ_N2    = A256((size_t)N_NODES * OUT_CH * 4);
constexpr size_t SZ_E     = A256((size_t)N_EDGES * 4);
constexpr size_t SZ_N     = A256((size_t)N_NODES * 4);

constexpr size_t OFF_Q1    = 0;
constexpr size_t OFF_K1    = OFF_Q1    + SZ_NH;
constexpr size_t OFF_V1    = OFF_K1    + SZ_NH;
constexpr size_t OFF_S1    = OFF_V1    + SZ_NH;
constexpr size_t OFF_AGG1  = OFF_S1    + SZ_NH;   // becomes h after combine+relu
constexpr size_t OFF_EXPS  = OFF_AGG1  + SZ_NH;
constexpr size_t OFF_MAXE  = OFF_EXPS  + SZ_E;
constexpr size_t OFF_SUME  = OFF_MAXE  + SZ_N;
constexpr size_t OFF_Q2    = OFF_SUME  + SZ_N;
constexpr size_t OFF_K2    = OFF_Q2    + SZ_N2;
constexpr size_t OFF_V2    = OFF_K2    + SZ_N2;
constexpr size_t OFF_S2    = OFF_V2    + SZ_N2;
constexpr size_t OFF_AGG2  = OFF_S2    + SZ_N2;
constexpr size_t OFF_MAXE2 = OFF_AGG2  + SZ_N2;
constexpr size_t OFF_SUME2 = OFF_MAXE2 + SZ_N;
constexpr size_t OFF_EXPS2 = OFF_SUME2 + SZ_N;
constexpr size_t OFF_AP    = OFF_EXPS2 + SZ_E;              // packed bf16 A (x)
constexpr size_t SZ_AP     = (size_t)MT * KS1 * 32 * 32;    // 25.6 MB
constexpr size_t OFF_BP    = OFF_AP + SZ_AP;                // packed bf16 B (4 weights)
constexpr size_t SZ_BP     = (size_t)4 * NT1 * KS1 * 32 * 32;

// ---------------- helpers ----------------
__device__ __forceinline__ unsigned short f2bf(float f) {
    unsigned u = __float_as_uint(f);
    unsigned r = u + 0x7FFFu + ((u >> 16) & 1u);   // RNE
    return (unsigned short)(r >> 16);
}
__device__ __forceinline__ unsigned fenc(float f) {      // monotonic float->uint
    unsigned u = __float_as_uint(f);
    return (u & 0x80000000u) ? ~u : (u | 0x80000000u);
}
__device__ __forceinline__ float fdec(unsigned u) {
    return __uint_as_float((u & 0x80000000u) ? (u ^ 0x80000000u) : ~u);
}

// ---------------- kernels ----------------
__global__ void zero_kernel(uint32_t* p, int n) {
    int i = blockIdx.x * blockDim.x + threadIdx.x;
    if (i < n) p[i] = 0u;
}

// Pack x[50000,256] f32 into bf16 WMMA A-layout (16-bit A 16x32 tile layout).
__global__ void pack_a_kernel(const float* __restrict__ x, uint32_t* __restrict__ ap) {
    int t = blockIdx.x * blockDim.x + threadIdx.x;      // MT*KS1*32 = 800000
    if (t >= MT * KS1 * 32) return;
    int lane = t & 31;
    int s    = (t >> 5) & 7;
    int mt   = t >> 8;
    int m  = mt * 16 + (lane & 15);
    int hi = lane >> 4;
    const float* xr = x + (size_t)m * IN_CH + s * 32 + hi * 8;
    uint32_t* dst = ap + (size_t)t * 8;                 // t == ((mt*KS1+s)*32+lane)
    float r0[8], r1[8];
#pragma unroll
    for (int i = 0; i < 8; i++) { r0[i] = xr[i]; r1[i] = xr[16 + i]; }
#pragma unroll
    for (int j = 0; j < 4; j++)
        dst[j] = (uint32_t)f2bf(r0[2 * j]) | ((uint32_t)f2bf(r0[2 * j + 1]) << 16);
#pragma unroll
    for (int j = 0; j < 4; j++)
        dst[4 + j] = (uint32_t)f2bf(r1[2 * j]) | ((uint32_t)f2bf(r1[2 * j + 1]) << 16);
}

// Pack the 4 weight matrices [256,200] f32 into bf16 WMMA B-layout (32x16 tiles,
// lane = column, K 2-per-dword, lane half h holds K 8h..8h+7 then 16+8h..16+8h+7).
__global__ void pack_b_kernel(const float* __restrict__ w0, const float* __restrict__ w1,
                              const float* __restrict__ w2, const float* __restrict__ w3,
                              uint32_t* __restrict__ bp) {
    int t = blockIdx.x * blockDim.x + threadIdx.x;      // 4*NT1*KS1*32 = 13312
    if (t >= 4 * NT1 * KS1 * 32) return;
    int lane = t & 31;
    int idx  = t >> 5;
    int s    = idx & 7;  idx >>= 3;
    int nt   = idx % NT1;
    int w    = idx / NT1;
    const float* W = (w == 0) ? w0 : (w == 1) ? w1 : (w == 2) ? w2 : w3;
    int n  = nt * 16 + (lane & 15);
    int hi = lane >> 4;
    int k0 = s * 32 + hi * 8;
    uint32_t* dst = bp + (size_t)(((w * NT1 + nt) * KS1 + s) * 32 + lane) * 8;
#pragma unroll
    for (int j = 0; j < 8; j++) {
        float f0 = (n < HID) ? W[(size_t)(k0 + 2 * j + (j >> 2) * 8) * HID + n]     : 0.f;
        float f1 = (n < HID) ? W[(size_t)(k0 + 2 * j + (j >> 2) * 8 + 1) * HID + n] : 0.f;
        dst[j] = (uint32_t)f2bf(f0) | ((uint32_t)f2bf(f1) << 16);
    }
}

// Layer-1 fused quad-GEMM: one wave per 16x16 tile of one of {q1,k1,v1,s1}.
__global__ void __launch_bounds__(256)
gemm1_kernel(const uint8_t* __restrict__ apb, const uint8_t* __restrict__ bpb,
             float* __restrict__ q1, float* __restrict__ k1,
             float* __restrict__ v1, float* __restrict__ s1,
             const float* __restrict__ bq, const float* __restrict__ bk,
             const float* __restrict__ bv, const float* __restrict__ bs) {
    int wid  = (blockIdx.x * blockDim.x + threadIdx.x) >> 5;
    int lane = threadIdx.x & 31;
    const int TOT = 4 * MT * NT1;
    if (wid >= TOT) return;
    int w   = wid / (MT * NT1);
    int rem = wid % (MT * NT1);
    int mt  = rem / NT1;
    int nt  = rem % NT1;
    const v16bf* A = (const v16bf*)apb + (size_t)mt * KS1 * 32 + lane;
    const v16bf* B = (const v16bf*)bpb + (size_t)(w * NT1 + nt) * KS1 * 32 + lane;
    v8f acc = {0.f, 0.f, 0.f, 0.f, 0.f, 0.f, 0.f, 0.f};
#pragma unroll
    for (int s = 0; s < KS1; s++) {
        v16bf a = A[(size_t)s * 32];
        v16bf b = B[(size_t)s * 32];
        if (s + 1 < KS1) {
            __builtin_prefetch(&A[(size_t)(s + 1) * 32], 0, 3);
            __builtin_prefetch(&B[(size_t)(s + 1) * 32], 0, 3);
        }
        acc = __builtin_amdgcn_wmma_f32_16x16x32_bf16(
            false, a, false, b, (short)0, acc, false, false);
    }
    float*       outp  = (w == 0) ? q1 : (w == 1) ? k1 : (w == 2) ? v1 : s1;
    const float* biasp = (w == 0) ? bq : (w == 1) ? bk : (w == 2) ? bv : bs;
    int n  = nt * 16 + (lane & 15);
    int hi = lane >> 4;
    if (n < HID) {
        float b  = biasp[n];
        int   m0 = mt * 16 + hi * 8;
#pragma unroll
        for (int r = 0; r < 8; r++)
            outp[(size_t)(m0 + r) * HID + n] = acc[r] + b;
    }
}

// scores[e] = dot(q[dst], k[src]) / sqrt(HID); running segment-max via atomicMax.
__global__ void score1_kernel(const int* __restrict__ ei,
                              const float* __restrict__ q1, const float* __restrict__ k1,
                              float* __restrict__ sco, uint32_t* __restrict__ maxe) {
    int e    = (blockIdx.x * blockDim.x + threadIdx.x) >> 5;
    int lane = threadIdx.x & 31;
    if (e >= N_EDGES) return;
    int s = ei[e];
    int d = ei[N_EDGES + e];
    const float* qr = q1 + (size_t)d * HID;
    const float* kr = k1 + (size_t)s * HID;
    float acc = 0.f;
    for (int c = lane; c < HID; c += 32) acc += qr[c] * kr[c];
#pragma unroll
    for (int off = 16; off > 0; off >>= 1) acc += __shfl_xor(acc, off, 32);
    if (lane == 0) {
        float sc = acc * 0.07071067811865475f;   // 1/sqrt(200)
        sco[e] = sc;
        atomicMax(maxe + d, fenc(sc));
    }
}

// e = exp(score - max[dst]); segment sum via atomicAdd. (used for both layers)
__global__ void expsum_kernel(const int* __restrict__ ei, const float* __restrict__ sco,
                              const uint32_t* __restrict__ maxe,
                              float* __restrict__ exps, float* __restrict__ sume) {
    int e = blockIdx.x * blockDim.x + threadIdx.x;
    if (e >= N_EDGES) return;
    int d = ei[N_EDGES + e];
    float m  = fdec(maxe[d]);
    float ex = __expf(sco[e] - m);
    exps[e] = ex;
    atomicAdd(sume + d, ex);
}

// agg[dst] += (exp/sum) * v[src]  (wave per edge, lane-strided channels)
__global__ void agg1_kernel(const int* __restrict__ ei, const float* __restrict__ exps,
                            const float* __restrict__ sume, const float* __restrict__ v1,
                            float* __restrict__ agg) {
    int e    = (blockIdx.x * blockDim.x + threadIdx.x) >> 5;
    int lane = threadIdx.x & 31;
    if (e >= N_EDGES) return;
    int s = ei[e];
    int d = ei[N_EDGES + e];
    float wgt = exps[e] / sume[d];
    const float* vr = v1 + (size_t)s * HID;
    float* ar = agg + (size_t)d * HID;
    for (int c = lane; c < HID; c += 32) atomicAdd(ar + c, wgt * vr[c]);
}

__global__ void combine1_kernel(float* __restrict__ agg, const float* __restrict__ s1) {
    int i = blockIdx.x * blockDim.x + threadIdx.x;
    if (i >= N_NODES * HID) return;
    float v = agg[i] + s1[i];
    agg[i] = v > 0.f ? v : 0.f;
}

// Layer-2 quad linear, OUT_CH=2: one thread per node.
__global__ void lin2_kernel(const float* __restrict__ h,
                            const float* __restrict__ Wq, const float* __restrict__ bq,
                            const float* __restrict__ Wk, const float* __restrict__ bk,
                            const float* __restrict__ Wv, const float* __restrict__ bv,
                            const float* __restrict__ Ws, const float* __restrict__ bs,
                            float* __restrict__ q2, float* __restrict__ k2,
                            float* __restrict__ v2, float* __restrict__ s2) {
    int i = blockIdx.x * blockDim.x + threadIdx.x;
    if (i >= N_NODES) return;
    const float* hr = h + (size_t)i * HID;
    float aq0 = 0, aq1 = 0, ak0 = 0, ak1 = 0, av0 = 0, av1 = 0, as0 = 0, as1 = 0;
    for (int c = 0; c < HID; c++) {
        float hv = hr[c];
        aq0 += hv * Wq[2 * c]; aq1 += hv * Wq[2 * c + 1];
        ak0 += hv * Wk[2 * c]; ak1 += hv * Wk[2 * c + 1];
        av0 += hv * Wv[2 * c]; av1 += hv * Wv[2 * c + 1];
        as0 += hv * Ws[2 * c]; as1 += hv * Ws[2 * c + 1];
    }
    q2[2 * i] = aq0 + bq[0]; q2[2 * i + 1] = aq1 + bq[1];
    k2[2 * i] = ak0 + bk[0]; k2[2 * i + 1] = ak1 + bk[1];
    v2[2 * i] = av0 + bv[0]; v2[2 * i + 1] = av1 + bv[1];
    s2[2 * i] = as0 + bs[0]; s2[2 * i + 1] = as1 + bs[1];
}

__global__ void score2_kernel(const int* __restrict__ ei,
                              const float* __restrict__ q2, const float* __restrict__ k2,
                              float* __restrict__ sco, uint32_t* __restrict__ maxe) {
    int e = blockIdx.x * blockDim.x + threadIdx.x;
    if (e >= N_EDGES) return;
    int s = ei[e];
    int d = ei[N_EDGES + e];
    float sc = (q2[2 * d] * k2[2 * s] + q2[2 * d + 1] * k2[2 * s + 1]) * 0.7071067811865476f;
    sco[e] = sc;
    atomicMax(maxe + d, fenc(sc));
}

__global__ void agg2_kernel(const int* __restrict__ ei, const float* __restrict__ exps,
                            const float* __restrict__ sume, const float* __restrict__ v2,
                            float* __restrict__ agg) {
    int e = blockIdx.x * blockDim.x + threadIdx.x;
    if (e >= N_EDGES) return;
    int s = ei[e];
    int d = ei[N_EDGES + e];
    float wgt = exps[e] / sume[d];
    atomicAdd(agg + 2 * d,     wgt * v2[2 * s]);
    atomicAdd(agg + 2 * d + 1, wgt * v2[2 * s + 1]);
}

__global__ void final_kernel(const float* __restrict__ agg2, const float* __restrict__ s2,
                             float* __restrict__ out) {
    int i = blockIdx.x * blockDim.x + threadIdx.x;
    if (i >= N_NODES * OUT_CH) return;
    float v = agg2[i] + s2[i];
    out[i] = v > 0.f ? v : 0.f;
}

// ---------------- launch ----------------
extern "C" void kernel_launch(void* const* d_in, const int* in_sizes, int n_in,
                              void* d_out, int out_size, void* d_ws, size_t ws_size,
                              hipStream_t stream) {
    (void)in_sizes; (void)n_in; (void)out_size; (void)ws_size;
    const float* x   = (const float*)d_in[0];
    const int*   ei  = (const int*)d_in[1];
    const float* Wq1 = (const float*)d_in[2];  const float* bq1 = (const float*)d_in[3];
    const float* Wk1 = (const float*)d_in[4];  const float* bk1 = (const float*)d_in[5];
    const float* Wv1 = (const float*)d_in[6];  const float* bv1 = (const float*)d_in[7];
    const float* Ws1 = (const float*)d_in[8];  const float* bs1 = (const float*)d_in[9];
    const float* Wq2 = (const float*)d_in[10]; const float* bq2 = (const float*)d_in[11];
    const float* Wk2 = (const float*)d_in[12]; const float* bk2 = (const float*)d_in[13];
    const float* Wv2 = (const float*)d_in[14]; const float* bv2 = (const float*)d_in[15];
    const float* Ws2 = (const float*)d_in[16]; const float* bs2 = (const float*)d_in[17];

    uint8_t* ws = (uint8_t*)d_ws;
    float*    q1    = (float*)(ws + OFF_Q1);
    float*    k1    = (float*)(ws + OFF_K1);
    float*    v1    = (float*)(ws + OFF_V1);
    float*    s1    = (float*)(ws + OFF_S1);
    float*    agg1  = (float*)(ws + OFF_AGG1);   // h after combine
    float*    exps  = (float*)(ws + OFF_EXPS);
    uint32_t* maxe  = (uint32_t*)(ws + OFF_MAXE);
    float*    sume  = (float*)(ws + OFF_SUME);
    float*    q2    = (float*)(ws + OFF_Q2);
    float*    k2    = (float*)(ws + OFF_K2);
    float*    v2    = (float*)(ws + OFF_V2);
    float*    s2    = (float*)(ws + OFF_S2);
    float*    agg2  = (float*)(ws + OFF_AGG2);
    uint32_t* maxe2 = (uint32_t*)(ws + OFF_MAXE2);
    float*    sume2 = (float*)(ws + OFF_SUME2);
    float*    exps2 = (float*)(ws + OFF_EXPS2);
    uint8_t*  ap    = ws + OFF_AP;
    uint8_t*  bp    = ws + OFF_BP;

    const int B = 256;
    auto blk = [](int n, int b) { return (n + b - 1) / b; };

    // zero accumulators / segment state
    zero_kernel<<<blk(N_NODES, B), B, 0, stream>>>(maxe, N_NODES);
    zero_kernel<<<blk(N_NODES, B), B, 0, stream>>>((uint32_t*)sume, N_NODES);
    zero_kernel<<<blk(N_NODES * HID, B), B, 0, stream>>>((uint32_t*)agg1, N_NODES * HID);
    zero_kernel<<<blk(N_NODES, B), B, 0, stream>>>(maxe2, N_NODES);
    zero_kernel<<<blk(N_NODES, B), B, 0, stream>>>((uint32_t*)sume2, N_NODES);
    zero_kernel<<<blk(N_NODES * OUT_CH, B), B, 0, stream>>>((uint32_t*)agg2, N_NODES * OUT_CH);

    // bf16 packing for WMMA
    pack_a_kernel<<<blk(MT * KS1 * 32, B), B, 0, stream>>>(x, (uint32_t*)ap);
    pack_b_kernel<<<blk(4 * NT1 * KS1 * 32, B), B, 0, stream>>>(Wq1, Wk1, Wv1, Ws1, (uint32_t*)bp);

    // layer-1 fused quad GEMM (q1,k1,v1,s1)
    {
        int waves = 4 * MT * NT1;
        gemm1_kernel<<<blk(waves * 32, B), B, 0, stream>>>(ap, bp, q1, k1, v1, s1,
                                                           bq1, bk1, bv1, bs1);
    }

    // layer-1 attention
    score1_kernel<<<blk(N_EDGES * 32, B), B, 0, stream>>>(ei, q1, k1, exps /*reuse as score buf? no*/, maxe);
    // NOTE: scores stored in exps buffer region? keep separate: reuse exps for scores then overwrite
    // (score1 wrote raw scores into `exps`; expsum reads them and rewrites with exp values)
    expsum_kernel<<<blk(N_EDGES, B), B, 0, stream>>>(ei, exps, maxe, exps, sume);
    agg1_kernel<<<blk(N_EDGES * 32, B), B, 0, stream>>>(ei, exps, sume, v1, agg1);
    combine1_kernel<<<blk(N_NODES * HID, B), B, 0, stream>>>(agg1, s1);

    // layer-2 linear + attention
    lin2_kernel<<<blk(N_NODES, B), B, 0, stream>>>(agg1, Wq2, bq2, Wk2, bk2, Wv2, bv2,
                                                   Ws2, bs2, q2, k2, v2, s2);
    score2_kernel<<<blk(N_EDGES, B), B, 0, stream>>>(ei, q2, k2, exps2, maxe2);
    expsum_kernel<<<blk(N_EDGES, B), B, 0, stream>>>(ei, exps2, maxe2, exps2, sume2);
    agg2_kernel<<<blk(N_EDGES, B), B, 0, stream>>>(ei, exps2, sume2, v2, agg2);
    final_kernel<<<blk(N_NODES * OUT_CH, B), B, 0, stream>>>(agg2, s2, (float*)d_out);
}
